// ROIAwareGNN_77498389889097
// MI455X (gfx1250) — compile-verified, matrix operand order checked by reading, not run on hardware
//
#include <hip/hip_runtime.h>
#include <hip/hip_bf16.h>
#include <stdint.h>

#define F_DIM 256
#define H_DIM 256
#define C_COM 7
#define G_NUM 16
#define BINS  4096
#define CDIV(a,b) (((a)+(b)-1)/(b))

typedef __attribute__((ext_vector_type(2)))  float    v2f;
typedef __attribute__((ext_vector_type(4)))  float    v4f;
typedef __attribute__((ext_vector_type(8)))  float    v8f;
typedef __attribute__((ext_vector_type(16))) _Float16 v16h;

#if defined(__gfx1250__) && __has_builtin(__builtin_amdgcn_wmma_f32_16x16x4_f32)
#define WMMA_F32_NATIVE 1
#else
#define WMMA_F32_NATIVE 0
#endif

// LDS B-strip layout: element (k,c) at ((k>>1)*64 + c)*2 + (k&1)
// -> the pair {B[k][c], B[k+1][c]} (k even) is one aligned 8-byte ds_load_b64,
//    and adjacent lanes are 2 banks apart (conflict-free).
__device__ __forceinline__ int lds_idx(int k, int c) {
    return ((((k >> 1) << 6) + c) << 1) + (k & 1);
}

// Stage a 256x64 B strip into LDS with the pair-interleave permutation.
// On gfx1250: per-lane-addressed async copy (global_load_async_to_lds_b32),
// LDS written directly, tracked by ASYNCcnt -> s_wait_asynccnt before barrier.
__device__ __forceinline__ void stage_B_strip(float* __restrict__ ldsB,
                                              const float* __restrict__ B,
                                              int cb, int tid) {
#if WMMA_F32_NATIVE
    #pragma unroll 4
    for (int i = tid; i < F_DIM * 64; i += 256) {
        const int kr = i >> 6, kc = i & 63;
        const float*   g = B + (size_t)kr * H_DIM + cb + kc;
        const unsigned l = (unsigned)(uintptr_t)&ldsB[lds_idx(kr, kc)]; // LDS byte address (low 32 bits)
        asm volatile("global_load_async_to_lds_b32 %0, %1, off"
                     :: "v"(l), "v"(g) : "memory");
    }
    asm volatile("s_wait_asynccnt 0x0" ::: "memory");
#else
    for (int i = tid; i < F_DIM * 64; i += 256) {
        const int kr = i >> 6, kc = i & 63;
        ldsB[lds_idx(kr, kc)] = B[(size_t)kr * H_DIM + cb + kc];
    }
#endif
}

// ---------------------------------------------------------------- utilities
__global__ void fill_f32_k(float* p, float v, int n) {
    int i = blockIdx.x * blockDim.x + threadIdx.x;
    if (i < n) p[i] = v;
}
__global__ void fill_i32_k(int* p, int v, int n) {
    int i = blockIdx.x * blockDim.x + threadIdx.x;
    if (i < n) p[i] = v;
}

// ---------------------------------------------------------------- degree / norm
__global__ void degree_k(const int* __restrict__ dst, const float* __restrict__ ew,
                         float* __restrict__ deg, int E) {
    int e = blockIdx.x * blockDim.x + threadIdx.x;
    if (e < E) atomicAdd(&deg[dst[e]], ew[e]);
}
__global__ void dinv_k(const float* __restrict__ deg, float* __restrict__ dinv, int N) {
    int n = blockIdx.x * blockDim.x + threadIdx.x;
    if (n < N) {
        float d = deg[n];
        dinv[n] = (d > 0.0f) ? rsqrtf(d) : 0.0f;
    }
}

// ---------------------------------------------------------------- community bucketing
__global__ void comm_count_k(const int* __restrict__ community, int* __restrict__ cntv, int N) {
    int n = blockIdx.x * blockDim.x + threadIdx.x;
    if (n < N) atomicAdd(&cntv[community[n]], 1);
}
__global__ void comm_scan_k(int* cntv, int* padOff) {
    int off = 0;
    for (int c = 0; c < C_COM; ++c) {
        padOff[c] = off;
        off += ((cntv[c] + 255) >> 8) << 8;   // pad each bucket to multiple of 256
        cntv[c] = 0;                           // reuse as scatter counters
    }
    padOff[C_COM] = off;
}
__global__ void comm_scatter_k(const int* __restrict__ community, const int* __restrict__ padOff,
                               int* __restrict__ ctr, int* __restrict__ perm, int N) {
    int n = blockIdx.x * blockDim.x + threadIdx.x;
    if (n >= N) return;
    int c = community[n];
    int slot = atomicAdd(&ctr[c], 1);
    perm[padOff[c] + slot] = n;
}

// ---------------------------------------------------------------- WMMA GEMM: C = A @ B (A: MxF, B: FxH)
// 256 threads = 8 waves; block async-stages a 256x64 B strip in LDS (64 KB);
// each wave computes a 32x64 output tile (8 f32 accumulators).
__global__ __launch_bounds__(256)
void gemm_dense_wmma(const float* __restrict__ A, const float* __restrict__ B,
                     float* __restrict__ C, int M) {
    __shared__ float ldsB[F_DIM * 64];
    const int tid = threadIdx.x;
    const int cb  = blockIdx.y << 6;
    stage_B_strip(ldsB, B, cb, tid);
    __syncthreads();

    const int wave = tid >> 5, lane = tid & 31;
    const int half = lane >> 4, lm = lane & 15;
    const int row0 = ((blockIdx.x << 3) + wave) << 5;
    if (row0 >= M) return;

    const int r0 = row0 + lm, r1 = row0 + 16 + lm;
    const float s0 = (r0 < M) ? 1.0f : 0.0f;
    const float s1 = (r1 < M) ? 1.0f : 0.0f;
    const float* A0 = A + (size_t)(r0 < M ? r0 : 0) * F_DIM;
    const float* A1 = A + (size_t)(r1 < M ? r1 : 0) * F_DIM;

    v8f acc[8] = {};
#if WMMA_F32_NATIVE
    for (int k0 = 0; k0 < F_DIM; k0 += 4) {
        const int ka = k0 + (half << 1);
        v2f a0 = *(const v2f*)(A0 + ka) * s0;        // global_load_b64 + v_pk_mul_f32
        v2f a1 = *(const v2f*)(A1 + ka) * s1;
        const float* Bk = ldsB + lds_idx(ka, lm);
        v2f b0 = *(const v2f*)(Bk);                  // ds_load_2addr_b64 (merged pairs)
        v2f b1 = *(const v2f*)(Bk + 32);
        v2f b2 = *(const v2f*)(Bk + 64);
        v2f b3 = *(const v2f*)(Bk + 96);
        acc[0] = __builtin_amdgcn_wmma_f32_16x16x4_f32(false, a0, false, b0, (short)0, acc[0], false, false);
        acc[1] = __builtin_amdgcn_wmma_f32_16x16x4_f32(false, a0, false, b1, (short)0, acc[1], false, false);
        acc[2] = __builtin_amdgcn_wmma_f32_16x16x4_f32(false, a0, false, b2, (short)0, acc[2], false, false);
        acc[3] = __builtin_amdgcn_wmma_f32_16x16x4_f32(false, a0, false, b3, (short)0, acc[3], false, false);
        acc[4] = __builtin_amdgcn_wmma_f32_16x16x4_f32(false, a1, false, b0, (short)0, acc[4], false, false);
        acc[5] = __builtin_amdgcn_wmma_f32_16x16x4_f32(false, a1, false, b1, (short)0, acc[5], false, false);
        acc[6] = __builtin_amdgcn_wmma_f32_16x16x4_f32(false, a1, false, b2, (short)0, acc[6], false, false);
        acc[7] = __builtin_amdgcn_wmma_f32_16x16x4_f32(false, a1, false, b3, (short)0, acc[7], false, false);
    }
#else
    for (int k0 = 0; k0 < F_DIM; k0 += 32) {
        const int kb = k0 + (half << 3);
        v16h a0, a1;
        #pragma unroll
        for (int j = 0; j < 8; ++j) {
            a0[j]     = (_Float16)(A0[kb + j] * s0);
            a0[j + 8] = (_Float16)(A0[kb + 16 + j] * s0);
            a1[j]     = (_Float16)(A1[kb + j] * s1);
            a1[j + 8] = (_Float16)(A1[kb + 16 + j] * s1);
        }
        #pragma unroll
        for (int t = 0; t < 4; ++t) {
            const int c = (t << 4) + lm;
            v16h b;
            #pragma unroll
            for (int j = 0; j < 8; ++j) {
                b[j]     = (_Float16)ldsB[lds_idx(kb + j, c)];
                b[j + 8] = (_Float16)ldsB[lds_idx(kb + 16 + j, c)];
            }
            acc[t]     = __builtin_amdgcn_wmma_f32_16x16x32_f16(false, a0, false, b, (short)0, acc[t], false, false);
            acc[4 + t] = __builtin_amdgcn_wmma_f32_16x16x32_f16(false, a1, false, b, (short)0, acc[4 + t], false, false);
        }
    }
#endif
    #pragma unroll
    for (int r = 0; r < 8; ++r) {
        int orow = row0 + r + (half << 3);
        if (orow < M) {
            float* Cr = C + (size_t)orow * H_DIM + cb + lm;
            Cr[0] = acc[0][r]; Cr[16] = acc[1][r]; Cr[32] = acc[2][r]; Cr[48] = acc[3][r];
        }
        orow += 16;
        if (orow < M) {
            float* Cr = C + (size_t)orow * H_DIM + cb + lm;
            Cr[0] = acc[4][r]; Cr[16] = acc[5][r]; Cr[32] = acc[6][r]; Cr[48] = acc[7][r];
        }
    }
}

// ---------------------------------------------------------------- WMMA GEMM: C[perm] = x[perm] @ basis[comm]
// Buckets are padded to multiples of 256, so each block is community-uniform.
__global__ __launch_bounds__(256)
void gemm_basis_wmma(const float* __restrict__ X, const float* __restrict__ basis,
                     const int* __restrict__ community, const int* __restrict__ perm,
                     float* __restrict__ C) {
    __shared__ float ldsB[F_DIM * 64];
    __shared__ int   sComm;
    const int tid = threadIdx.x;
    const int cb  = blockIdx.y << 6;
    const int blockRow0 = blockIdx.x << 8;

    if (tid == 0) sComm = -1;
    __syncthreads();
    {
        int pn = perm[blockRow0 + tid];
        if (pn >= 0) sComm = community[pn];   // benign same-value race
    }
    __syncthreads();
    if (sComm < 0) return;                    // fully padded block

    stage_B_strip(ldsB, basis + (size_t)sComm * F_DIM * H_DIM, cb, tid);
    __syncthreads();

    const int wave = tid >> 5, lane = tid & 31;
    const int half = lane >> 4, lm = lane & 15;
    const int row0 = blockRow0 + (wave << 5);

    const int n0 = perm[row0 + lm];
    const int n1 = perm[row0 + 16 + lm];
    const float s0 = (n0 >= 0) ? 1.0f : 0.0f;
    const float s1 = (n1 >= 0) ? 1.0f : 0.0f;
    const float* A0 = X + (size_t)(n0 >= 0 ? n0 : 0) * F_DIM;
    const float* A1 = X + (size_t)(n1 >= 0 ? n1 : 0) * F_DIM;

    v8f acc[8] = {};
#if WMMA_F32_NATIVE
    for (int k0 = 0; k0 < F_DIM; k0 += 4) {
        const int ka = k0 + (half << 1);
        v2f a0 = *(const v2f*)(A0 + ka) * s0;
        v2f a1 = *(const v2f*)(A1 + ka) * s1;
        const float* Bk = ldsB + lds_idx(ka, lm);
        v2f b0 = *(const v2f*)(Bk);
        v2f b1 = *(const v2f*)(Bk + 32);
        v2f b2 = *(const v2f*)(Bk + 64);
        v2f b3 = *(const v2f*)(Bk + 96);
        acc[0] = __builtin_amdgcn_wmma_f32_16x16x4_f32(false, a0, false, b0, (short)0, acc[0], false, false);
        acc[1] = __builtin_amdgcn_wmma_f32_16x16x4_f32(false, a0, false, b1, (short)0, acc[1], false, false);
        acc[2] = __builtin_amdgcn_wmma_f32_16x16x4_f32(false, a0, false, b2, (short)0, acc[2], false, false);
        acc[3] = __builtin_amdgcn_wmma_f32_16x16x4_f32(false, a0, false, b3, (short)0, acc[3], false, false);
        acc[4] = __builtin_amdgcn_wmma_f32_16x16x4_f32(false, a1, false, b0, (short)0, acc[4], false, false);
        acc[5] = __builtin_amdgcn_wmma_f32_16x16x4_f32(false, a1, false, b1, (short)0, acc[5], false, false);
        acc[6] = __builtin_amdgcn_wmma_f32_16x16x4_f32(false, a1, false, b2, (short)0, acc[6], false, false);
        acc[7] = __builtin_amdgcn_wmma_f32_16x16x4_f32(false, a1, false, b3, (short)0, acc[7], false, false);
    }
#else
    for (int k0 = 0; k0 < F_DIM; k0 += 32) {
        const int kb = k0 + (half << 3);
        v16h a0, a1;
        #pragma unroll
        for (int j = 0; j < 8; ++j) {
            a0[j]     = (_Float16)(A0[kb + j] * s0);
            a0[j + 8] = (_Float16)(A0[kb + 16 + j] * s0);
            a1[j]     = (_Float16)(A1[kb + j] * s1);
            a1[j + 8] = (_Float16)(A1[kb + 16 + j] * s1);
        }
        #pragma unroll
        for (int t = 0; t < 4; ++t) {
            const int c = (t << 4) + lm;
            v16h b;
            #pragma unroll
            for (int j = 0; j < 8; ++j) {
                b[j]     = (_Float16)ldsB[lds_idx(kb + j, c)];
                b[j + 8] = (_Float16)ldsB[lds_idx(kb + 16 + j, c)];
            }
            acc[t]     = __builtin_amdgcn_wmma_f32_16x16x32_f16(false, a0, false, b, (short)0, acc[t], false, false);
            acc[4 + t] = __builtin_amdgcn_wmma_f32_16x16x32_f16(false, a1, false, b, (short)0, acc[4 + t], false, false);
        }
    }
#endif
    #pragma unroll
    for (int r = 0; r < 8; ++r) {
        int sr = perm[row0 + r + (half << 3)];
        if (sr >= 0) {
            float* Cr = C + (size_t)sr * H_DIM + cb + lm;
            Cr[0] = acc[0][r]; Cr[16] = acc[1][r]; Cr[32] = acc[2][r]; Cr[48] = acc[3][r];
        }
        sr = perm[row0 + 16 + r + (half << 3)];
        if (sr >= 0) {
            float* Cr = C + (size_t)sr * H_DIM + cb + lm;
            Cr[0] = acc[4][r]; Cr[16] = acc[5][r]; Cr[32] = acc[6][r]; Cr[48] = acc[7][r];
        }
    }
}

// ---------------------------------------------------------------- GCN aggregation
// self-loop contribution: out[n] = dinv[n]^2 * m[n]   (vectorized float4)
__global__ void agg_init_k(const float* __restrict__ m, const float* __restrict__ dinv,
                           float* __restrict__ out, int NH4) {
    int i = blockIdx.x * blockDim.x + threadIdx.x;
    if (i < NH4) {
        int n = i >> 6;
        float di = dinv[n];
        ((v4f*)out)[i] = ((const v4f*)m)[i] * (di * di);
    }
}
// one wave per edge; float4 gathers (global_load_b128) + f32 atomic scatter (L2-resident)
__global__ __launch_bounds__(256)
void edge_agg_k(const int* __restrict__ src, const int* __restrict__ dst,
                const float* __restrict__ ew, const float* __restrict__ dinv,
                const float* __restrict__ m, float* __restrict__ out, int E) {
    int wid  = (blockIdx.x * blockDim.x + threadIdx.x) >> 5;
    int lane = threadIdx.x & 31;
    if (wid >= E) return;
    int s = src[wid], d = dst[wid];
    float nrm = dinv[s] * ew[wid] * dinv[d];
    const v4f* ms4 = (const v4f*)(m + (size_t)s * H_DIM);
    float*     od  = out + (size_t)d * H_DIM;
    __builtin_prefetch(od + (lane << 2), 1, 3);    // global_prefetch_b8
    v4f lo = ms4[lane]      * nrm;
    v4f hi = ms4[lane + 32] * nrm;
    #pragma unroll
    for (int c = 0; c < 4; ++c) {
        atomicAdd(&od[(lane << 2) + c], lo[c]);
        atomicAdd(&od[128 + (lane << 2) + c], hi[c]);
    }
}
__global__ void relu_bias_k(const float* __restrict__ agg, const float* __restrict__ bias,
                            float* __restrict__ h, int NH4) {
    int i = blockIdx.x * blockDim.x + threadIdx.x;
    if (i < NH4) {
        v4f v = ((const v4f*)agg)[i] + ((const v4f*)bias)[i & 63];
        #pragma unroll
        for (int c = 0; c < 4; ++c) v[c] = fmaxf(v[c], 0.0f);
        ((v4f*)h)[i] = v;
    }
}

// ---------------------------------------------------------------- pooling scores + histogram
__global__ __launch_bounds__(256)
void score_k(const float* __restrict__ h, const float* __restrict__ pw, const float* __restrict__ pb,
             float* __restrict__ scores, int* __restrict__ hist, int N) {
    int node = (blockIdx.x * blockDim.x + threadIdx.x) >> 5;
    int lane = threadIdx.x & 31;
    if (node >= N) return;
    const float* hr = h + (size_t)node * H_DIM;
    float sum = 0.0f;
    #pragma unroll
    for (int j = 0; j < 8; ++j) {
        int f = lane + (j << 5);
        sum += hr[f] * pw[f];
    }
    #pragma unroll
    for (int off = 16; off > 0; off >>= 1) sum += __shfl_xor(sum, off, 32);
    if (lane == 0) {
        float sg = 1.0f / (1.0f + __expf(-(sum + pb[0])));
        scores[node] = sg;
        int bin = (int)(sg * (float)BINS);
        bin = min(max(bin, 0), BINS - 1);
        atomicAdd(&hist[bin], 1);
    }
}
__global__ void thresh_k(const int* __restrict__ hist, int* __restrict__ sel, int K) {
    int cum = 0, t = 0, chi = 0, need = K;
    for (int b = BINS - 1; b >= 0; --b) {
        int c = hist[b];
        if (cum + c >= K) { t = b; chi = cum; need = K - cum; break; }
        cum += c;
    }
    sel[0] = t; sel[1] = chi; sel[2] = need; sel[3] = 0; sel[4] = 0;
}
__global__ void select_k(const float* __restrict__ scores, int* __restrict__ sel,
                         int* __restrict__ top, int N) {
    int n = blockIdx.x * blockDim.x + threadIdx.x;
    if (n >= N) return;
    float sg = scores[n];
    int bin = (int)(sg * (float)BINS);
    bin = min(max(bin, 0), BINS - 1);
    int t = sel[0];
    if (bin > t) {
        int p = atomicAdd(&sel[3], 1);
        top[p] = n;
    } else if (bin == t) {
        int q = atomicAdd(&sel[4], 1);
        if (q < sel[2]) top[sel[1] + q] = n;
    }
}
__global__ __launch_bounds__(256)
void pool_k(const int* __restrict__ top, const int* __restrict__ batch,
            const float* __restrict__ h, float* __restrict__ sums, float* __restrict__ cnt, int K) {
    int i    = (blockIdx.x * blockDim.x + threadIdx.x) >> 5;
    int lane = threadIdx.x & 31;
    if (i >= K) return;
    int node = top[i];
    int g = batch[node];
    if (lane == 0) atomicAdd(&cnt[g], 1.0f);
    const float* hr = h + (size_t)node * H_DIM;
    float* sr = sums + (size_t)g * H_DIM;
    #pragma unroll
    for (int j = 0; j < 8; ++j) {
        int f = lane + (j << 5);
        atomicAdd(&sr[f], hr[f]);
    }
}
// tiny tail MLP: mean pool -> fc1+relu -> fc2 -> out[16,2]
__global__ __launch_bounds__(256)
void final_k(const float* __restrict__ sums, const float* __restrict__ cnt,
             const float* __restrict__ fc1w, const float* __restrict__ fc1b,
             const float* __restrict__ fc2w, const float* __restrict__ fc2b,
             float* __restrict__ dout) {
    __shared__ float hg[G_NUM * H_DIM];
    __shared__ float z[G_NUM * 128];
    int tid = threadIdx.x;
    for (int i = tid; i < G_NUM * H_DIM; i += 256) {
        int g = i >> 8;
        hg[i] = sums[i] / fmaxf(cnt[g], 1.0f);
    }
    __syncthreads();
    for (int i = tid; i < G_NUM * 128; i += 256) {
        int g = i >> 7, o = i & 127;
        float acc = fc1b[o];
        for (int f = 0; f < H_DIM; ++f) acc += hg[(g << 8) + f] * fc1w[f * 128 + o];
        z[i] = fmaxf(acc, 0.0f);
    }
    __syncthreads();
    if (tid < 32) {
        int g = tid >> 1, o = tid & 1;
        float acc = fc2b[o];
        for (int f = 0; f < 128; ++f) acc += z[(g << 7) + f] * fc2w[f * 2 + o];
        dout[g * 2 + o] = acc;
    }
}
__global__ void topidx_k(const int* __restrict__ top, float* __restrict__ dout, int K) {
    int i = blockIdx.x * blockDim.x + threadIdx.x;
    if (i < K) dout[i] = (float)top[i];
}

// ---------------------------------------------------------------- driver
extern "C" void kernel_launch(void* const* d_in, const int* in_sizes, int n_in,
                              void* d_out, int out_size, void* d_ws, size_t ws_size,
                              hipStream_t stream) {
    (void)n_in; (void)out_size; (void)ws_size;
    const int N    = in_sizes[0] / F_DIM;
    const int E    = in_sizes[4];
    const int KSEL = (N / 2 > 0) ? (N / 2) : 1;
    const int NH   = N * H_DIM;

    const float* x         = (const float*)d_in[0];
    const int*   src       = (const int*)d_in[1];
    const int*   dst       = src + E;
    const int*   batch     = (const int*)d_in[2];
    const int*   community = (const int*)d_in[3];
    const float* ew        = (const float*)d_in[4];
    const float* basis     = (const float*)d_in[5];
    const float* W1        = (const float*)d_in[6];
    const float* b1        = (const float*)d_in[7];
    const float* W2        = (const float*)d_in[8];
    const float* b2        = (const float*)d_in[9];
    const float* pw        = (const float*)d_in[10];
    const float* pb        = (const float*)d_in[11];
    const float* fc1w      = (const float*)d_in[12];
    const float* fc1b      = (const float*)d_in[13];
    const float* fc2w      = (const float*)d_in[14];
    const float* fc2b      = (const float*)d_in[15];
    float* out = (float*)d_out;

    // workspace bump allocation
    const int permBlocks = CDIV(N, 256) + C_COM;       // basis-GEMM blocks (row dim)
    const int permSize   = permBlocks * 256;
    char* p = (char*)d_ws;
    auto take = [&](size_t bytes) { char* r = p; p += (bytes + 255) & ~(size_t)255; return r; };
    float* bufH   = (float*)take((size_t)NH * 4);
    float* bufM   = (float*)take((size_t)NH * 4);
    float* bufO   = (float*)take((size_t)NH * 4);
    float* deg    = (float*)take((size_t)N * 4);
    float* dinv   = (float*)take((size_t)N * 4);
    float* scores = (float*)take((size_t)N * 4);
    int*   perm   = (int*)take((size_t)permSize * 4);
    int*   ccnt   = (int*)take(64);
    int*   padOff = (int*)take(64);
    int*   hist   = (int*)take((size_t)BINS * 4);
    int*   sel    = (int*)take(64);
    int*   top    = (int*)take((size_t)KSEL * 4);
    float* sums   = (float*)take((size_t)G_NUM * H_DIM * 4);
    float* cnt    = (float*)take((size_t)G_NUM * 4);

    // init
    fill_f32_k<<<CDIV(N, 256), 256, 0, stream>>>(deg, 1.0f, N);   // self-loop weight
    fill_i32_k<<<CDIV(permSize, 256), 256, 0, stream>>>(perm, -1, permSize);
    fill_i32_k<<<1, 32, 0, stream>>>(ccnt, 0, 8);
    fill_i32_k<<<CDIV(BINS, 256), 256, 0, stream>>>(hist, 0, BINS);
    fill_f32_k<<<CDIV(G_NUM * H_DIM, 256), 256, 0, stream>>>(sums, 0.0f, G_NUM * H_DIM);
    fill_f32_k<<<1, 32, 0, stream>>>(cnt, 0.0f, G_NUM);

    // degree + symmetric norm
    degree_k<<<CDIV(E, 256), 256, 0, stream>>>(dst, ew, deg, E);
    dinv_k<<<CDIV(N, 256), 256, 0, stream>>>(deg, dinv, N);

    // bucket nodes by community (pad buckets to multiples of 256)
    comm_count_k<<<CDIV(N, 256), 256, 0, stream>>>(community, ccnt, N);
    comm_scan_k<<<1, 1, 0, stream>>>(ccnt, padOff);
    comm_scatter_k<<<CDIV(N, 256), 256, 0, stream>>>(community, padOff, ccnt, perm, N);

    // MoE basis GEMM (community-uniform blocks via padded perm): bufH = x @ basis[comm]
    gemm_basis_wmma<<<dim3(permBlocks, H_DIM / 64), 256, 0, stream>>>(x, basis, community, perm, bufH);

    const int rblocks = CDIV(N, 256);
    // GCN layer 1
    gemm_dense_wmma<<<dim3(rblocks, H_DIM / 64), 256, 0, stream>>>(bufH, W1, bufM, N);
    agg_init_k<<<CDIV(NH / 4, 256), 256, 0, stream>>>(bufM, dinv, bufO, NH / 4);
    edge_agg_k<<<CDIV(E, 8), 256, 0, stream>>>(src, dst, ew, dinv, bufM, bufO, E);
    relu_bias_k<<<CDIV(NH / 4, 256), 256, 0, stream>>>(bufO, b1, bufH, NH / 4);
    // GCN layer 2
    gemm_dense_wmma<<<dim3(rblocks, H_DIM / 64), 256, 0, stream>>>(bufH, W2, bufM, N);
    agg_init_k<<<CDIV(NH / 4, 256), 256, 0, stream>>>(bufM, dinv, bufO, NH / 4);
    edge_agg_k<<<CDIV(E, 8), 256, 0, stream>>>(src, dst, ew, dinv, bufM, bufO, E);
    relu_bias_k<<<CDIV(NH / 4, 256), 256, 0, stream>>>(bufO, b2, bufH, NH / 4);

    // top-k pooling (histogram threshold select), segment mean, tail MLP
    score_k<<<CDIV(N, 8), 256, 0, stream>>>(bufH, pw, pb, scores, hist, N);
    thresh_k<<<1, 1, 0, stream>>>(hist, sel, KSEL);
    select_k<<<CDIV(N, 256), 256, 0, stream>>>(scores, sel, top, N);
    pool_k<<<CDIV(KSEL, 8), 256, 0, stream>>>(top, batch, bufH, sums, cnt, KSEL);
    final_k<<<1, 256, 0, stream>>>(sums, cnt, fc1w, fc1b, fc2w, fc2b, out);
    topidx_k<<<CDIV(KSEL, 256), 256, 0, stream>>>(top, out + G_NUM * 2, KSEL);
}
